// CosineKMeans_80702435491911
// MI455X (gfx1250) — matrix operand order repaired and checked by reading, not compiled, not run on hardware
//
#include <hip/hip_runtime.h>
#include <hip/hip_bf16.h>

typedef __attribute__((ext_vector_type(16))) __bf16 v16bf;
typedef __attribute__((ext_vector_type(8)))  float  v8f;

#define NROWS 1000000
#define DIM   64
#define KC    256
#define RPB   256            // rows per block (8 waves x 32 rows)

// d_ws layout (floats): sums[KC*DIM] | counts[KC] | simsum[1]
#define WS_COUNTS (KC * DIM)
#define WS_SIMSUM (KC * DIM + KC)

__global__ __launch_bounds__(256)
void kmeans_assign(const float* __restrict__ X, const float* __restrict__ C,
                   float* __restrict__ labels_out, float* __restrict__ sums,
                   float* __restrict__ counts, float* __restrict__ simsum)
{
    // B fragments for all of C, bf16, pre-swizzled: 16 ntiles x 2 ksteps x (32 lanes x 16 elems)
    __shared__ __bf16 cb[KC * DIM];          // 32 KB
    __shared__ int    labLDS[8][32];

    const int tid = threadIdx.x;

    // ---- stage C -> LDS in WMMA B-fragment order (bf16) ----
    // s = ((nt*2+ks)*32 + l)*16 + e ; value = C[nt*16 + (l&15)][ks*32 + (l>=16?8:0) + (e&7) + (e>=8?16:0)]
    #pragma unroll
    for (int i = 0; i < 16; ++i) {
        int s    = tid * 64 + i * 4;
        int e    = s & 15;
        int l    = (s >> 4) & 31;
        int frag = s >> 9;
        int nt = frag >> 1, ks = frag & 1;
        int n = nt * 16 + (l & 15);
        int k = ks * 32 + ((l >= 16) ? 8 : 0) + (e & 7) + ((e >= 8) ? 16 : 0);
        float4 v = *(const float4*)(C + n * DIM + k);
        union { __bf16 h[4]; uint2 u; } pk;
        pk.h[0] = (__bf16)v.x; pk.h[1] = (__bf16)v.y;
        pk.h[2] = (__bf16)v.z; pk.h[3] = (__bf16)v.w;
        *(uint2*)(&cb[s]) = pk.u;
    }

    // ---- per wave: 32 rows = two 16-row A tiles sharing the same B fragments ----
    const int wave    = tid >> 5;
    const int lane    = tid & 31;
    const int rowBase = blockIdx.x * RPB + wave * 32;
    const int b       = (lane >= 16) ? 8 : 0;           // half-row column base
    const int offs[8] = {0, 4, 16, 20, 32, 36, 48, 52};

    float4 rr[2][8];
    float  inv[2];
    bool   rowValid[2];
    v16bf  A[2][2];

    #pragma unroll
    for (int t = 0; t < 2; ++t) {
        int myRow   = rowBase + t * 16 + (lane & 15);
        rowValid[t] = (myRow < NROWS);
        int rsafe   = rowValid[t] ? myRow : (NROWS - 1);
        const float* xr = X + (size_t)rsafe * DIM + b;
        float ss = 0.f;
        #pragma unroll
        for (int i = 0; i < 8; ++i) {
            rr[t][i] = *(const float4*)(xr + offs[i]);
            ss += rr[t][i].x * rr[t][i].x + rr[t][i].y * rr[t][i].y
                + rr[t][i].z * rr[t][i].z + rr[t][i].w * rr[t][i].w;
        }
        ss += __shfl_xor(ss, 16, 32);                   // combine half-rows
        inv[t] = rsqrtf(fmaxf(ss, 1e-24f));             // 1 / max(||x||, 1e-12)
        #pragma unroll
        for (int i = 0; i < 4; ++i) {
            float4 a = rr[t][i];
            A[t][0][4*i+0] = (__bf16)(a.x * inv[t]); A[t][0][4*i+1] = (__bf16)(a.y * inv[t]);
            A[t][0][4*i+2] = (__bf16)(a.z * inv[t]); A[t][0][4*i+3] = (__bf16)(a.w * inv[t]);
            float4 c = rr[t][i + 4];
            A[t][1][4*i+0] = (__bf16)(c.x * inv[t]); A[t][1][4*i+1] = (__bf16)(c.y * inv[t]);
            A[t][1][4*i+2] = (__bf16)(c.z * inv[t]); A[t][1][4*i+3] = (__bf16)(c.w * inv[t]);
        }
    }

    __syncthreads();

    // ---- WMMA over 16 centroid tiles, software-pipelined B fragments ----
    float bestV[2][8];
    int   bestN[2][8];
    #pragma unroll
    for (int t = 0; t < 2; ++t)
        #pragma unroll
        for (int r = 0; r < 8; ++r) { bestV[t][r] = -2.0f; bestN[t][r] = 0; }

    const int nsel = lane & 15;
    const __bf16* fbase = &cb[lane * 16];
    v16bf B0 = *(const v16bf*)(fbase);
    v16bf B1 = *(const v16bf*)(fbase + 512);

    // Partial unroll: enough ILP/pipelining per body, but bounded register
    // pressure (keep the kernel under 256 VGPRs -> no s_set_vgpr_msb churn,
    // higher occupancy for the memory-bound phases).
    #pragma unroll 4
    for (int nt = 0; nt < 16; ++nt) {
        int ntn = (nt + 1) & 15;                        // branchless wrap prefetch
        v16bf nB0 = *(const v16bf*)(fbase + ntn * 1024);
        v16bf nB1 = *(const v16bf*)(fbase + ntn * 1024 + 512);

        v8f acc0 = {}, acc1 = {};
        acc0 = __builtin_amdgcn_wmma_f32_16x16x32_bf16(false, A[0][0], false, B0, (short)0, acc0, false, false);
        acc1 = __builtin_amdgcn_wmma_f32_16x16x32_bf16(false, A[1][0], false, B0, (short)0, acc1, false, false);
        acc0 = __builtin_amdgcn_wmma_f32_16x16x32_bf16(false, A[0][1], false, B1, (short)0, acc0, false, false);
        acc1 = __builtin_amdgcn_wmma_f32_16x16x32_bf16(false, A[1][1], false, B1, (short)0, acc1, false, false);

        int n = nt * 16 + nsel;
        #pragma unroll
        for (int r = 0; r < 8; ++r) {
            float v0 = acc0[r];
            if (v0 > bestV[0][r]) { bestV[0][r] = v0; bestN[0][r] = n; }
            float v1 = acc1[r];
            if (v1 > bestV[1][r]) { bestV[1][r] = v1; bestN[1][r] = n; }
        }
        B0 = nB0; B1 = nB1;
    }

    // butterfly argmax across the 16 lanes sharing each row (first-index tie-break)
    #pragma unroll
    for (int m = 1; m < 16; m <<= 1) {
        #pragma unroll
        for (int t = 0; t < 2; ++t)
            #pragma unroll
            for (int r = 0; r < 8; ++r) {
                float ov = __shfl_xor(bestV[t][r], m, 32);
                int   on = __shfl_xor(bestN[t][r], m, 32);
                if (ov > bestV[t][r] || (ov == bestV[t][r] && on < bestN[t][r])) {
                    bestV[t][r] = ov; bestN[t][r] = on;
                }
            }
    }

    // lanes (l&15)==0 own final results: lower half -> rows 0..7, upper -> rows 8..15
    if ((lane & 15) == 0) {
        int mbase = (lane >> 4) * 8;
        float lsum = 0.f;
        #pragma unroll
        for (int t = 0; t < 2; ++t)
            #pragma unroll
            for (int r = 0; r < 8; ++r) {
                labLDS[wave][t * 16 + mbase + r] = bestN[t][r];
                int row = rowBase + t * 16 + mbase + r;
                if (row < NROWS) {
                    labels_out[row] = (float)bestN[t][r];
                    lsum += bestV[t][r];
                    atomicAdd(&counts[bestN[t][r]], 1.0f);
                }
            }
        atomicAdd(simsum, lsum);
    }
    __syncthreads();

    // ---- scatter-add normalized rows into sums[label] ----
    #pragma unroll
    for (int t = 0; t < 2; ++t) {
        int myLab = labLDS[wave][t * 16 + (lane & 15)];
        if (rowValid[t]) {
            float* srow = sums + myLab * DIM + b;
            #pragma unroll
            for (int i = 0; i < 8; ++i) {
                atomicAdd(&srow[offs[i] + 0], rr[t][i].x * inv[t]);
                atomicAdd(&srow[offs[i] + 1], rr[t][i].y * inv[t]);
                atomicAdd(&srow[offs[i] + 2], rr[t][i].z * inv[t]);
                atomicAdd(&srow[offs[i] + 3], rr[t][i].w * inv[t]);
            }
        }
    }
}

__global__ __launch_bounds__(32)
void kmeans_update(const float* __restrict__ C, const float* __restrict__ sums,
                   const float* __restrict__ counts, const float* __restrict__ simsum,
                   float* __restrict__ out)
{
    const int k = blockIdx.x;
    const int t = threadIdx.x;
    float cnt = counts[k];
    float v0 = sums[k * DIM + t];
    float v1 = sums[k * DIM + t + 32];
    float o0, o1;
    if (cnt > 0.f) {
        float m0 = v0 / cnt, m1 = v1 / cnt;
        float ss = m0 * m0 + m1 * m1;
        #pragma unroll
        for (int m = 1; m < 32; m <<= 1) ss += __shfl_xor(ss, m, 32);
        float inv = rsqrtf(fmaxf(ss, 1e-24f));
        o0 = m0 * inv; o1 = m1 * inv;
    } else {
        o0 = C[k * DIM + t];
        o1 = C[k * DIM + t + 32];
    }
    out[k * DIM + t]      = o0;
    out[k * DIM + t + 32] = o1;
    if (k == 0 && t == 0)
        out[KC * DIM + NROWS] = 1.0f - simsum[0] / (float)NROWS;
}

extern "C" void kernel_launch(void* const* d_in, const int* in_sizes, int n_in,
                              void* d_out, int out_size, void* d_ws, size_t ws_size,
                              hipStream_t stream)
{
    const float* X = (const float*)d_in[0];
    const float* C = (const float*)d_in[1];
    float* out     = (float*)d_out;

    float* sums   = (float*)d_ws;
    float* counts = (float*)d_ws + WS_COUNTS;
    float* ssum   = (float*)d_ws + WS_SIMSUM;

    hipMemsetAsync(d_ws, 0, (size_t)(KC * DIM + KC + 1) * sizeof(float), stream);

    float* labels = out + KC * DIM;          // tuple order: C_new, labels, mean_J
    int nblocks = (NROWS + RPB - 1) / RPB;
    kmeans_assign<<<nblocks, 256, 0, stream>>>(X, C, labels, sums, counts, ssum);
    kmeans_update<<<KC, 32, 0, stream>>>(C, sums, counts, ssum, out);
}
